// NTXentLoss_18451179503736
// MI455X (gfx1250) — matrix operand order repaired and checked by reading, not compiled, not run on hardware
//
#include <hip/hip_runtime.h>
#include <hip/hip_bf16.h>
#include <math.h>

// ---------------- problem constants ----------------
constexpr int N_PAIRS = 4096;
constexpr int ROWS    = 8192;    // 2N
constexpr int D       = 256;
#define TEMP_INV 2.0f            // 1 / 0.5

typedef __attribute__((ext_vector_type(16))) _Float16 v16h;
typedef __attribute__((ext_vector_type(8)))  _Float16 v8h;
typedef __attribute__((ext_vector_type(8)))  float    v8f;
typedef __attribute__((ext_vector_type(4)))  unsigned int u32x4;
typedef __attribute__((ext_vector_type(8)))  int      i32x8;
typedef __attribute__((ext_vector_type(4)))  int      i32x4;

#if defined(__has_builtin)
#if __has_builtin(__builtin_amdgcn_tensor_load_to_lds) && __has_builtin(__builtin_amdgcn_s_wait_tensorcnt)
#define HAVE_TDM 1
#endif
#endif
#ifndef HAVE_TDM
#define HAVE_TDM 0
#endif

// ---------------- kernel 1: L2-normalize rows -> f16, zero rowsum ----------------
__global__ void nt_normalize(const float* __restrict__ zi, const float* __restrict__ zj,
                             _Float16* __restrict__ Rh, float* __restrict__ rowsum) {
  const int row = blockIdx.x;
  const int t   = threadIdx.x;
  const float* src = (row < N_PAIRS) ? (zi + (size_t)row * D)
                                     : (zj + (size_t)(row - N_PAIRS) * D);
  const float x = src[t];
  float ss = x * x;
#pragma unroll
  for (int off = 16; off >= 1; off >>= 1) ss += __shfl_xor(ss, off, 32);
  __shared__ float wsum[8];
  const int lane = t & 31, wid = t >> 5;
  if (lane == 0) wsum[wid] = ss;
  __syncthreads();
  __shared__ float sinv;
  if (t == 0) {
    float tot = 0.f;
#pragma unroll
    for (int i = 0; i < 8; ++i) tot += wsum[i];
    sinv = 1.0f / fmaxf(sqrtf(tot), 1e-12f);
    rowsum[row] = 0.0f;
  }
  __syncthreads();
  Rh[(size_t)row * D + t] = (_Float16)(x * sinv);
}

// ---------------- kernel 2: positives pos[i] = dot(Rh[i], Rh[i+N]) ----------------
__global__ void nt_positives(const _Float16* __restrict__ Rh, float* __restrict__ pos) {
  const int i = blockIdx.x, t = threadIdx.x;
  float p = (float)Rh[(size_t)i * D + t] * (float)Rh[(size_t)(i + N_PAIRS) * D + t];
#pragma unroll
  for (int off = 16; off >= 1; off >>= 1) p += __shfl_xor(p, off, 32);
  __shared__ float wsum[8];
  const int lane = t & 31, wid = t >> 5;
  if (lane == 0) wsum[wid] = p;
  __syncthreads();
  if (t == 0) {
    float tot = 0.f;
#pragma unroll
    for (int k = 0; k < 8; ++k) tot += wsum[k];
    pos[i] = tot;
  }
}

// ---------------- kernel 3: fused WMMA GEMM + row sums of exp(sim/T) ----------------
constexpr int ROWS_PER_BLOCK = 128;                  // 8 waves x 16 rows
constexpr int NSPLIT         = 8;                    // col-range split for parallelism
constexpr int COLS_PER_STEP  = 32;                   // 2 x 16-col subtiles per barrier step
constexpr int NSTEPS         = ROWS / COLS_PER_STEP / NSPLIT;  // 32 steps of 32 cols
constexpr int LDS_ROW_H      = 264;                  // 256 halves + 16B pad -> 528B stride
constexpr int BUF_H          = 32 * LDS_ROW_H;       // halves per LDS buffer

#if HAVE_TDM
// Tensor DMA Descriptor (D#, ISA cdna5 §8.3/8.4) for a 2D tile:
// 32 rows x 512 bytes (f16), row stride 512B in memory, LDS padded +16B / 512B.
__device__ inline void tdm_load_tile(unsigned lds_off, const _Float16* gptr) {
  const unsigned long long ga = (unsigned long long)(uintptr_t)gptr;
  u32x4 g0;
  g0[0] = 1u;                                          // count=1, user D#, dense
  g0[1] = lds_off;                                     // [63:32]  lds_addr (bytes)
  g0[2] = (unsigned)(ga & 0xFFFFFFFFu);                // [95:64]  global_addr lo
  g0[3] = (unsigned)((ga >> 32) & 0x1FFFFFFu) | (2u << 30); // addr[56:32] | type=2
  i32x8 g1;
  // data_size=1 (2B) | pad_enable | pad_interval=6 (128 dwords) | pad_amount=3 (4 dwords)
  g1[0] = (1 << 16) | (1 << 20) | (6 << 22) | (3 << 25);
  g1[1] = (int)(((unsigned)D & 0xFFFFu) << 16);        // tensor_dim0 lo16
  g1[2] = (int)((((unsigned)D >> 16) & 0xFFFFu) | (((unsigned)ROWS & 0xFFFFu) << 16));
  g1[3] = (int)((((unsigned)ROWS >> 16) & 0xFFFFu) | (((unsigned)D & 0xFFFFu) << 16)); // tile_dim0=256
  g1[4] = 32;                                          // tile_dim1=32, tile_dim2=0
  g1[5] = D;                                           // tensor_dim0_stride lo32
  g1[6] = 0;                                           // stride hi / dim1_stride lo
  g1[7] = 0;
  const i32x4 z4 = {0, 0, 0, 0};
  const i32x8 z8 = {0, 0, 0, 0, 0, 0, 0, 0};
  __builtin_amdgcn_tensor_load_to_lds(g0, g1, z4, z4, z8, 0);
}
#endif

__global__ void __launch_bounds__(256)
nt_simrow(const _Float16* __restrict__ Rh, float* __restrict__ rowsum) {
  __shared__ __align__(16) _Float16 Btile[2 * BUF_H];
  const int t      = threadIdx.x;
  const int wave   = t >> 5;
  const int lane   = t & 31;
  const int lane16 = lane & 15;
  const int laneHi = lane >> 4;
  const int rowBase = blockIdx.x * ROWS_PER_BLOCK + wave * 16;
  const int colBlock0 = blockIdx.y * (NSTEPS * COLS_PER_STEP);

  // ---- load this wave's A fragments (16 rows x 256 K) into registers ----
  // 16-bit A 16x32 layout: lane<16 holds row M=lane, K {0..7,16..23}; lane>=16 same row, K {8..15,24..31}
  v16h afrag[8];
  {
    const int arow = rowBase + lane16;
#pragma unroll
    for (int kb = 0; kb < 8; ++kb) {
      const _Float16* ap = Rh + (size_t)arow * D + kb * 32 + laneHi * 8;
      const v8h lo = *(const v8h*)ap;         // K offsets +0..7
      const v8h hi = *(const v8h*)(ap + 16);  // K offsets +16..23
#pragma unroll
      for (int i = 0; i < 8; ++i) { afrag[kb][i] = lo[i]; afrag[kb][i + 8] = hi[i]; }
    }
  }

  float rsum[8];
#pragma unroll
  for (int e = 0; e < 8; ++e) rsum[e] = 0.f;

#if HAVE_TDM
  // ---- software pipeline prologue: DMA step 0 into buffer 0 ----
  if (wave == 0)
    tdm_load_tile((unsigned)(uintptr_t)&Btile[0], Rh + (size_t)colBlock0 * D);
#endif

  for (int step = 0; step < NSTEPS; ++step) {
    const int buf = step & 1;
#if HAVE_TDM
    if (wave == 0) {
      if (step + 1 < NSTEPS) {
        // issue next tile's DMA into the other buffer, then wait only for the
        // oldest transfer (this step's) -- TDM ops from one wave finish in order
        tdm_load_tile((unsigned)(uintptr_t)&Btile[(buf ^ 1) * BUF_H],
                      Rh + (size_t)(colBlock0 + (step + 1) * COLS_PER_STEP) * D);
        __builtin_amdgcn_s_wait_tensorcnt((short)1);
      } else {
        __builtin_amdgcn_s_wait_tensorcnt((short)0);
      }
    }
    __syncthreads();   // buf[step&1] ready for all waves
#else
    {
      const int r = t >> 3;              // 32 rows
      const int c = (t & 7) * 32;        // 32 halves per thread
      const _Float16* g = Rh + (size_t)(colBlock0 + step * COLS_PER_STEP + r) * D + c;
#pragma unroll
      for (int q = 0; q < 4; ++q)
        *(v8h*)&Btile[buf * BUF_H + r * LDS_ROW_H + c + q * 8] = *(const v8h*)(g + q * 8);
    }
    __syncthreads();
#endif

    // ---- two 16x16 output subtiles, each a full-K chain of 8 WMMAs ----
#pragma unroll
    for (int sub = 0; sub < 2; ++sub) {
      // Load ALL B fragments first so the 8 WMMAs issue back-to-back.
      // 16-bit B 32x16 layout: lane<16 -> N=lane, K {0..7,16..23}; lane>=16 -> same N, K {8..15,24..31}
      v16h bfrag[8];
#pragma unroll
      for (int kb = 0; kb < 8; ++kb) {
        const _Float16* bp = &Btile[buf * BUF_H + (sub * 16 + lane16) * LDS_ROW_H +
                                    kb * 32 + laneHi * 8];
        const v8h b0 = *(const v8h*)bp;
        const v8h b1 = *(const v8h*)(bp + 16);
#pragma unroll
        for (int i = 0; i < 8; ++i) { bfrag[kb][i] = b0[i]; bfrag[kb][i + 8] = b1[i]; }
      }
      v8f c = {};
#pragma unroll
      for (int kb = 0; kb < 8; ++kb)
        c = __builtin_amdgcn_wmma_f32_16x16x32_f16(false, afrag[kb], false, bfrag[kb],
                                                   (short)0, c, false, false);

      // ---- epilogue: exp(sim/T) with diagonal masked, accumulate per row ----
      const int col  = colBlock0 + step * COLS_PER_STEP + sub * 16 + lane16;
      const int rowA = rowBase + laneHi * 8;  // C layout: vgpr e, lanes 0-15 -> M=e; 16-31 -> M=8+e
#pragma unroll
      for (int e = 0; e < 8; ++e) {
        const float ex = (col == rowA + e) ? 0.0f : __expf(c[e] * TEMP_INV);
        rsum[e] += ex;
      }
    }
    __syncthreads();   // all reads of buf done before it is re-filled at step+2
  }

  // ---- reduce across each 16-lane half (fixed N partition), add to rowsum ----
#pragma unroll
  for (int e = 0; e < 8; ++e) {
    float v = rsum[e];
#pragma unroll
    for (int off = 1; off < 16; off <<= 1) v += __shfl_xor(v, off, 32);
    if (lane16 == 0) atomicAdd(&rowsum[rowBase + laneHi * 8 + e], v);
  }
}

// ---------------- kernel 4: loss = mean(log(denom) - pos/T) ----------------
__global__ void nt_loss(const float* __restrict__ rowsum, const float* __restrict__ pos,
                        float* __restrict__ out) {
  const int t = threadIdx.x;
  float acc = 0.f;
  for (int r = t; r < ROWS; r += 256)
    acc += logf(rowsum[r]) - pos[r & (N_PAIRS - 1)] * TEMP_INV;
#pragma unroll
  for (int off = 16; off >= 1; off >>= 1) acc += __shfl_xor(acc, off, 32);
  __shared__ float wsum[8];
  const int lane = t & 31, wid = t >> 5;
  if (lane == 0) wsum[wid] = acc;
  __syncthreads();
  if (t == 0) {
    float tot = 0.f;
#pragma unroll
    for (int i = 0; i < 8; ++i) tot += wsum[i];
    out[0] = tot / (float)ROWS;
  }
}

// ---------------- host launcher ----------------
extern "C" void kernel_launch(void* const* d_in, const int* in_sizes, int n_in,
                              void* d_out, int out_size, void* d_ws, size_t ws_size,
                              hipStream_t stream) {
  (void)in_sizes; (void)n_in; (void)out_size; (void)ws_size;
  const float* zi = (const float*)d_in[0];
  const float* zj = (const float*)d_in[1];
  float* out = (float*)d_out;

  char* ws = (char*)d_ws;
  _Float16* Rh     = (_Float16*)ws;                                   // 4 MiB
  float*    rowsum = (float*)(ws + (size_t)ROWS * D * 2);             // 32 KiB
  float*    pos    = (float*)(ws + (size_t)ROWS * D * 2 + ROWS * 4);  // 16 KiB

  nt_normalize<<<ROWS, 256, 0, stream>>>(zi, zj, Rh, rowsum);
  nt_positives<<<N_PAIRS, 256, 0, stream>>>(Rh, pos);
  dim3 grid(ROWS / ROWS_PER_BLOCK, NSPLIT);
  nt_simrow<<<grid, 256, 0, stream>>>(Rh, rowsum);
  nt_loss<<<1, 256, 0, stream>>>(rowsum, pos, out);
}
